// Attention_89361089560777
// MI455X (gfx1250) — matrix operand order repaired and checked by reading, not compiled, not run on hardware
//
#include <hip/hip_runtime.h>
#include <stdint.h>
#include <stddef.h>

// ---------- types ----------
typedef __attribute__((ext_vector_type(16))) __bf16        v16bf;
typedef __attribute__((ext_vector_type(8)))  float         v8f;
typedef __attribute__((ext_vector_type(4)))  unsigned int  u32x4;
typedef __attribute__((ext_vector_type(2)))  unsigned int  u32x2;
typedef __attribute__((ext_vector_type(4)))  float         f32x4;

union FragU { u32x4 u[2]; v16bf v; };   // 32 bytes = 16 bf16 = one WMMA A/B operand

__device__ __forceinline__ unsigned int pack_bf16(float a, float b) {
  unsigned int ua = __float_as_uint(a); ua = (ua + 0x7FFFu + ((ua >> 16) & 1u)) >> 16;
  unsigned int ub = __float_as_uint(b); ub = (ub + 0x7FFFu + ((ub >> 16) & 1u)) >> 16;
  return (ua & 0xFFFFu) | (ub << 16);
}
__device__ __forceinline__ unsigned short f2bf(float a) {
  unsigned int ua = __float_as_uint(a);
  return (unsigned short)((ua + 0x7FFFu + ((ua >> 16) & 1u)) >> 16);
}

// ============================================================================
// Elementwise fp32 -> bf16 (row-major preserved). 8 elements / thread.
// ============================================================================
__global__ __launch_bounds__(256) void tobf16_kernel(
    const float* __restrict__ X, unsigned short* __restrict__ Y, int n8)
{
  const int i = blockIdx.x * 256 + threadIdx.x;
  if (i < n8) {
    const f32x4 x0 = *(const f32x4*)(X + (size_t)i * 8);
    const f32x4 x1 = *(const f32x4*)(X + (size_t)i * 8 + 4);
    u32x4 d;
    d.x = pack_bf16(x0.x, x0.y); d.y = pack_bf16(x0.z, x0.w);
    d.z = pack_bf16(x1.x, x1.y); d.w = pack_bf16(x1.z, x1.w);
    *(u32x4*)(Y + (size_t)i * 8) = d;
  }
}

// ============================================================================
// W [768k x 768n] fp32  ->  Wt [768n x 768k] bf16 (transpose via LDS tile)
// ============================================================================
__global__ __launch_bounds__(256) void wt_kernel(
    const float* __restrict__ W, unsigned short* __restrict__ Wt)
{
  __shared__ float tile[32][33];
  const int n0 = blockIdx.x * 32;
  const int k0 = blockIdx.y * 32;
  const int tx = threadIdx.x & 31, ty = threadIdx.x >> 5;  // ty 0..7
  for (int p = 0; p < 4; ++p) {
    const int kk = ty + p * 8;
    tile[kk][tx] = W[(size_t)(k0 + kk) * 768 + n0 + tx];
  }
  __syncthreads();
  const int r = threadIdx.x >> 3;   // 0..31 output row (n)
  const int c = threadIdx.x & 7;    // 0..7, 4 k-values each
  u32x2 d;
  d.x = pack_bf16(tile[4 * c + 0][r], tile[4 * c + 1][r]);
  d.y = pack_bf16(tile[4 * c + 2][r], tile[4 * c + 3][r]);
  *(u32x2*)(&Wt[(size_t)(n0 + r) * 768 + k0 + 4 * c]) = d;
}

// ============================================================================
// GEMM: C[8192,768] = A(bf16,row-major) * Wt(bf16, pre-transposed)^T + bias
// Block 256 threads = 8 waves arranged 4(M) x 2(N): BM=128, BN=128.
// Wave tile 32x64: 2 A-frags x 4 B-frags -> 8 WMMAs / 32-K step, double-buffered
// fragment loads straight from global (L2-resident operands, no LDS).
// modes: 0/1 = Q/K out [B,nh,S,d] bf16 ; 2 = V^T out [B,nh,d,S] bf16
//        3 = fp32 row-major output
// ============================================================================
struct Frags { FragU a[2]; FragU b[4]; };

__device__ __forceinline__ void load_frags(Frags& f, const char* abase,
                                           const char* bbase, int kbyte, int lhalf)
{
  #pragma unroll
  for (int i = 0; i < 2; ++i) {
    const char* p = abase + (size_t)i * 16 * 1536 + kbyte + lhalf * 16;
    f.a[i].u[0] = *(const u32x4*)p;           // global_load_b128
    f.a[i].u[1] = *(const u32x4*)(p + 32);
  }
  #pragma unroll
  for (int j = 0; j < 4; ++j) {
    const char* p = bbase + (size_t)j * 16 * 1536 + kbyte + lhalf * 16;
    f.b[j].u[0] = *(const u32x4*)p;
    f.b[j].u[1] = *(const u32x4*)(p + 32);
  }
}

__global__ __launch_bounds__(256) void gemm_kernel(
    const unsigned short* __restrict__ A,    // [8192,768] bf16
    const unsigned short* __restrict__ Wt,   // [768,768] bf16 (n-major)
    const float* __restrict__ bias,
    unsigned short* __restrict__ outb, float* __restrict__ outf, int mode)
{
  const int t = threadIdx.x;
  const int wave = t >> 5, lane = t & 31, lhalf = lane >> 4, l16 = lane & 15;
  const int wr = wave & 3, wc = wave >> 2;
  const int blockM = blockIdx.y * 128;
  const int blockN = blockIdx.x * 128;

  const char* abase = (const char*)A  + (size_t)(blockM + wr * 32 + l16) * 1536;
  const char* bbase = (const char*)Wt + (size_t)(blockN + wc * 64 + l16) * 1536;

  v8f acc[2][4];
  #pragma unroll
  for (int i = 0; i < 2; ++i)
    #pragma unroll
    for (int j = 0; j < 4; ++j)
      #pragma unroll
      for (int e = 0; e < 8; ++e) acc[i][j][e] = 0.0f;

  Frags f0, f1;
  load_frags(f0, abase, bbase, 0, lhalf);
  for (int kt = 0; kt < 24; ++kt) {
    Frags& cur = (kt & 1) ? f1 : f0;
    Frags& nxt = (kt & 1) ? f0 : f1;
    if (kt < 23) load_frags(nxt, abase, bbase, (kt + 1) * 64, lhalf);
    #pragma unroll
    for (int i = 0; i < 2; ++i)
      #pragma unroll
      for (int j = 0; j < 4; ++j)
        acc[i][j] = __builtin_amdgcn_wmma_f32_16x16x32_bf16(
            false, cur.a[i].v, false, cur.b[j].v, (short)0, acc[i][j], false, false);
  }

  // ---- epilogue: bias + per-mode scatter store ----
  #pragma unroll
  for (int i = 0; i < 2; ++i) {
    #pragma unroll
    for (int j = 0; j < 4; ++j) {
      const int colg = blockN + wc * 64 + j * 16 + l16;
      const float bv = bias[colg];
      #pragma unroll
      for (int r = 0; r < 8; ++r) {
        const int rowg = blockM + wr * 32 + i * 16 + r + 8 * lhalf;
        const float val = acc[i][j][r] + bv;
        if (mode == 3) {
          outf[(size_t)rowg * 768 + colg] = val;
        } else {
          const int b = rowg >> 10, s = rowg & 1023;
          const int head = colg >> 6, dd = colg & 63;
          size_t off;
          if (mode == 2) off = (((size_t)b * 12 + head) * 64 + dd) * 1024 + s;  // V^T
          else           off = (((size_t)b * 12 + head) * 1024 + s) * 64 + dd;  // Q,K
          outb[off] = f2bf(val);
        }
      }
    }
  }
}

// ============================================================================
// Attention: block per (b, head, 16-query tile). 256 threads = 8 waves.
// ============================================================================
__global__ __launch_bounds__(256) void attn_kernel(
    const unsigned short* __restrict__ q,   // [B,nh,S,d] bf16
    const unsigned short* __restrict__ k,   // [B,nh,S,d] bf16
    const unsigned short* __restrict__ vT,  // [B,nh,d,S] bf16
    const float* __restrict__ att_bias,     // [B,S,S,nh] fp32
    const int*  __restrict__ mask,          // [B,S,S]
    unsigned short* __restrict__ attn_out)  // [B,S,768] bf16
{
  extern __shared__ char smem[];
  float*          sS   = (float*)smem;                                   // 64 KB
  unsigned short* sP   = (unsigned short*)(smem + 16 * 1024 * 4);        // 32 KB
  float*          sRed = (float*)(smem + 16 * 1024 * 4 + 16 * 1024 * 2); // 1 KB

  const int t = threadIdx.x;
  const int wave = t >> 5, lane = t & 31, lhalf = lane >> 4, l16 = lane & 15;
  const int q0 = blockIdx.x * 16;
  const int head = blockIdx.y;
  const int b = blockIdx.z;

  const size_t bh = (size_t)b * 12 + head;
  const char* qbase = (const char*)(q  + bh * 1024 * 64);
  const char* kbase = (const char*)(k  + bh * 1024 * 64);
  const char* vbase = (const char*)(vT + bh * 64 * 1024);

  FragU aq[2];
  #pragma unroll
  for (int ks = 0; ks < 2; ++ks) {
    const char* p = qbase + (size_t)(q0 + l16) * 128 + ks * 64 + lhalf * 16;
    aq[ks].u[0] = *(const u32x4*)p;
    aq[ks].u[1] = *(const u32x4*)(p + 32);
  }

  // ---- phase 1: scores; wave covers keys [128*wave, 128*wave+128) ----
  const float scale = 0.125f;
  for (int j = 0; j < 8; ++j) {
    const int n0 = wave * 128 + j * 16;
    const int kg = n0 + l16;
    // prefetch next tile's bias/mask lines while WMMAs run
    if (j < 7) {
      const size_t nq = (size_t)b * 1024 + q0 + 8 * lhalf;
      __builtin_prefetch(&att_bias[(nq * 1024 + kg + 16) * 12 + head], 0, 0);
      __builtin_prefetch(&mask[nq * 1024 + kg + 16], 0, 0);
    }
    v8f c; for (int e = 0; e < 8; ++e) c[e] = 0.0f;
    #pragma unroll
    for (int ks = 0; ks < 2; ++ks) {
      FragU bk;
      const char* p = kbase + (size_t)(n0 + l16) * 128 + ks * 64 + lhalf * 16;
      bk.u[0] = *(const u32x4*)p;
      bk.u[1] = *(const u32x4*)(p + 32);
      c = __builtin_amdgcn_wmma_f32_16x16x32_bf16(
          false, aq[ks].v, false, bk.v, (short)0, c, false, false);
    }
    #pragma unroll
    for (int r = 0; r < 8; ++r) {
      const int m  = r + 8 * lhalf;
      const int qg = q0 + m;
      float sc = c[r] * scale +
                 att_bias[(((size_t)b * 1024 + qg) * 1024 + kg) * 12 + head];
      if (mask[((size_t)b * 1024 + qg) * 1024 + kg] != 0) sc = 1e-14f;
      sS[m * 1024 + kg] = sc;
    }
  }
  __syncthreads();

  // ---- phase 2: softmax, 16 threads per row ----
  {
    const int row = t >> 4, seg = t & 15;
    const float* rp = &sS[row * 1024 + seg * 64];
    float mx = -3.4e38f;
    for (int i = 0; i < 64; ++i) mx = fmaxf(mx, rp[i]);
    sRed[row * 16 + seg] = mx;
    __syncthreads();
    float rmax = sRed[row * 16];
    for (int i = 1; i < 16; ++i) rmax = fmaxf(rmax, sRed[row * 16 + i]);
    __syncthreads();
    float sum = 0.0f;
    for (int i = 0; i < 64; ++i) sum += __expf(rp[i] - rmax);
    sRed[row * 16 + seg] = sum;
    __syncthreads();
    float rsum = 0.0f;
    for (int i = 0; i < 16; ++i) rsum += sRed[row * 16 + i];
    const float inv = 1.0f / rsum;
    unsigned short* wp = &sP[row * 1024 + seg * 64];
    for (int i = 0; i < 64; ++i) wp[i] = f2bf(__expf(rp[i] - rmax) * inv);
  }
  __syncthreads();

  // ---- phase 3: P @ V, waves 0..3 each own 16 d-columns ----
  if (wave < 4) {
    v8f c; for (int e = 0; e < 8; ++e) c[e] = 0.0f;
    const char* vrow = vbase + (size_t)(wave * 16 + l16) * 2048;
    for (int ks = 0; ks < 32; ++ks) {
      FragU a, bb;
      const char* ap = (const char*)sP + (size_t)l16 * 2048 + ks * 64 + lhalf * 16;
      a.u[0] = *(const u32x4*)ap;            // ds_load_b128
      a.u[1] = *(const u32x4*)(ap + 32);
      const char* bp = vrow + ks * 64 + lhalf * 16;
      bb.u[0] = *(const u32x4*)bp;           // global_load_b128
      bb.u[1] = *(const u32x4*)(bp + 32);
      c = __builtin_amdgcn_wmma_f32_16x16x32_bf16(
          false, a.v, false, bb.v, (short)0, c, false, false);
    }
    #pragma unroll
    for (int r = 0; r < 8; ++r) {
      const int m  = r + 8 * lhalf;
      const int qg = q0 + m;
      attn_out[((size_t)b * 1024 + qg) * 768 + head * 64 + wave * 16 + l16] = f2bf(c[r]);
    }
  }
}

// ============================================================================
extern "C" void kernel_launch(void* const* d_in, const int* in_sizes, int n_in,
                              void* d_out, int out_size, void* d_ws, size_t ws_size,
                              hipStream_t stream) {
  const float* h        = (const float*)d_in[0];
  const float* att_bias = (const float*)d_in[1];
  const int*   mask     = (const int*)d_in[2];
  const float* Wq = (const float*)d_in[3];
  const float* bq = (const float*)d_in[4];
  const float* Wk = (const float*)d_in[5];
  const float* bk = (const float*)d_in[6];
  const float* Wv = (const float*)d_in[7];
  const float* bv = (const float*)d_in[8];
  const float* Wo = (const float*)d_in[9];
  const float* bo = (const float*)d_in[10];
  float* out = (float*)d_out;

  char* ws = (char*)d_ws;
  const size_t MB = 1024u * 1024u;
  unsigned short* qb  = (unsigned short*)(ws);            // 12 MB
  unsigned short* kb  = (unsigned short*)(ws + 12 * MB);  // 12 MB
  unsigned short* vtb = (unsigned short*)(ws + 24 * MB);  // 12 MB (V^T)
  unsigned short* ab  = (unsigned short*)(ws + 36 * MB);  // 12 MB attn out
  unsigned short* hb  = (unsigned short*)(ws + 48 * MB);  // 12 MB h bf16
  unsigned short* wtq = (unsigned short*)(ws + 60 * MB);  // 1.125 MB each
  unsigned short* wtk = (unsigned short*)(ws + 62 * MB);
  unsigned short* wtv = (unsigned short*)(ws + 64 * MB);
  unsigned short* wto = (unsigned short*)(ws + 66 * MB);

  dim3 block(256);

  // --- pre-convert operands to bf16 (h row-major; weights transposed) ---
  tobf16_kernel<<<dim3(3072), block, 0, stream>>>(h, hb, 8192 * 768 / 8);
  dim3 wgrid(24, 24);
  wt_kernel<<<wgrid, block, 0, stream>>>(Wq, wtq);
  wt_kernel<<<wgrid, block, 0, stream>>>(Wk, wtk);
  wt_kernel<<<wgrid, block, 0, stream>>>(Wv, wtv);
  wt_kernel<<<wgrid, block, 0, stream>>>(Wo, wto);

  // --- QKV projections ---
  dim3 ggrid(6, 64);  // N/128, M/128
  gemm_kernel<<<ggrid, block, 0, stream>>>(hb, wtq, bq, qb,  nullptr, 0);
  gemm_kernel<<<ggrid, block, 0, stream>>>(hb, wtk, bk, kb,  nullptr, 1);
  gemm_kernel<<<ggrid, block, 0, stream>>>(hb, wtv, bv, vtb, nullptr, 2);

  // --- attention ---
  dim3 agrid(64, 12, 8);
  size_t shmem = 16 * 1024 * 4 + 16 * 1024 * 2 + 16 * 16 * 4;  // 97 KB
  attn_kernel<<<agrid, block, shmem, stream>>>(qb, kb, vtb, att_bias, mask, ab);

  // --- output projection ---
  gemm_kernel<<<ggrid, block, 0, stream>>>(ab, wto, bo, nullptr, out, 3);
}
